// LSTMModel_4166118277587
// MI455X (gfx1250) — matrix operand order repaired and checked by reading, not compiled, not run on hardware
//
#include <hip/hip_runtime.h>
#include <hip/hip_bf16.h>

typedef __attribute__((ext_vector_type(16))) _Float16 v16h;
typedef __attribute__((ext_vector_type(8)))  _Float16 v8h;
typedef __attribute__((ext_vector_type(8)))  float    v8f;

#define INPUT_SZ 21
#define HIDDEN   50
#define OUT_N    24
#define BATCH    4096
#define SEQ      336
#define G4H      200   // 4*HIDDEN
#define NT       13    // n-tiles: 200 padded to 208
#define NWAVE    4
#define BLOCK    (NWAVE * 32)
#define NXS      3     // ceil(16*21 / BLOCK) x-staging slots per thread
#define NEW      7     // ceil(16*50 / BLOCK) elementwise slots per thread

__device__ __forceinline__ float fast_tanh(float x) {
#if __has_builtin(__builtin_amdgcn_tanhf)
    return __builtin_amdgcn_tanhf(x);      // gfx1250 v_tanh_f32 (1 trans op)
#else
    return 1.0f - 2.0f / (__expf(2.0f * x) + 1.0f);
#endif
}
__device__ __forceinline__ float fast_sigmoid(float x) {
    return 0.5f * fast_tanh(0.5f * x) + 0.5f;
}
__device__ __forceinline__ v16h cat16(v8h lo, v8h hi) {
    v16h r;
    #pragma unroll
    for (int i = 0; i < 8; ++i) { r[i] = lo[i]; r[i + 8] = hi[i]; }
    return r;
}

// One gate tile: acc = bias ; acc += xA*WihB[nt] ; acc += hA*WhhB[nt,ks=0..1]
__device__ __forceinline__ v8f gate_tile(const v16h& a_x, const v16h& a_h0,
                                         const v16h& a_h1,
                                         const _Float16* __restrict__ bih,
                                         const _Float16* __restrict__ bhh,
                                         float bvq)
{
    v8f acc;
    #pragma unroll
    for (int v = 0; v < 8; ++v) acc[v] = bvq;
    const v16h b0 = *(const v16h*)bih;
    acc = __builtin_amdgcn_wmma_f32_16x16x32_f16(false, a_x,  false, b0,
                                                 (short)0, acc, false, false);
    const v16h b1 = *(const v16h*)bhh;
    acc = __builtin_amdgcn_wmma_f32_16x16x32_f16(false, a_h0, false, b1,
                                                 (short)0, acc, false, false);
    const v16h b2 = *(const v16h*)(bhh + 32 * 16);   // second K-step tile
    acc = __builtin_amdgcn_wmma_f32_16x16x32_f16(false, a_h1, false, b2,
                                                 (short)0, acc, false, false);
    return acc;
}

// 4 waves cooperate on one 16-row batch tile. Weights live in LDS as f16 WMMA
// B-fragments. Per timestep: gates[16x208] = bias + xA*WihB + hA*WhhB via
// v_wmma_f32_16x16x32_f16 (tiles wid,wid+4,wid+8 straight-line; tile 12 on
// wave 0), then the LSTM cell update (v_tanh_f32) overlapped with staging the
// next timestep's x. Two barriers per step; all index math hoisted.
__global__ __launch_bounds__(BLOCK)
void lstm_wmma_kernel(const float* __restrict__ x,
                      const float* __restrict__ W_ih,
                      const float* __restrict__ W_hh,
                      const float* __restrict__ b_ih,
                      const float* __restrict__ b_hh,
                      const float* __restrict__ W_fc,
                      const float* __restrict__ b_fc,
                      float* __restrict__ out)
{
    __shared__ __align__(32) _Float16 s_wihB[NT * 32 * 16];      // 13,312 B
    __shared__ __align__(32) _Float16 s_whhB[NT * 2 * 32 * 16];  // 26,624 B
    __shared__ __align__(32) _Float16 s_xA[16 * 32];             //  1,024 B
    __shared__ __align__(32) _Float16 s_hA[16 * 64];             //  2,048 B
    __shared__ __align__(32) float    s_gates[16 * G4H];         // 12,800 B
    __shared__ __align__(32) float    s_c[16 * HIDDEN];          //  3,200 B
    __shared__ __align__(32) float    s_h32[16 * HIDDEN];        //  3,200 B
    __shared__ __align__(32) float    s_bias[NT * 16];           //    832 B
                                                                 // 63,040 B

    const int tid   = threadIdx.x;
    const int lane  = tid & 31;
    const int wid   = tid >> 5;    // wave id 0..3
    const int half  = lane >> 4;   // K-half selector (B frag) / row-half (C frag)
    const int nloc  = lane & 15;   // column within tile / A row
    const int bbase = blockIdx.x * 16;

    // ---- Stage W_ih into B-fragment layout: B[k][n] = W_ih[n][k], K 21->32 ----
    #pragma unroll 1
    for (int t = wid; t < NT; t += NWAVE) {
        const int n  = t * 16 + nloc;
        const int kb = half * 16;
        _Float16* dst = &s_wihB[(t * 32 + lane) * 16];
        #pragma unroll
        for (int v = 0; v < 16; ++v) {
            const int k = kb + v;
            const float w = (n < G4H && k < INPUT_SZ) ? W_ih[n * INPUT_SZ + k] : 0.0f;
            dst[v] = (_Float16)w;
        }
    }
    // ---- W_hh: B[k][n] = W_hh[n][k], K 50->64 (two K-steps per tile) ----
    #pragma unroll 1
    for (int tt = wid; tt < NT * 2; tt += NWAVE) {
        const int t = tt >> 1, ks = tt & 1;
        const int n  = t * 16 + nloc;
        const int kb = ks * 32 + half * 16;
        _Float16* dst = &s_whhB[(tt * 32 + lane) * 16];
        #pragma unroll
        for (int v = 0; v < 16; ++v) {
            const int k = kb + v;
            const float w = (n < G4H && k < HIDDEN) ? W_hh[n * HIDDEN + k] : 0.0f;
            dst[v] = (_Float16)w;
        }
    }
    // Combined gate bias per column (padded with zeros to 208)
    for (int i = tid; i < NT * 16; i += BLOCK)
        s_bias[i] = (i < G4H) ? (b_ih[i] + b_hh[i]) : 0.0f;

    // Zero state: h (f16 incl. K-pad), c, x K-pad columns
    for (int i = tid; i < 16 * 64; i += BLOCK) s_hA[i] = (_Float16)0.0f;
    for (int i = tid; i < 16 * HIDDEN; i += BLOCK) s_c[i] = 0.0f;
    for (int i = tid; i < 16 * 32; i += BLOCK) s_xA[i] = (_Float16)0.0f;

    // ---- Hoisted per-thread index math (time-invariant) ----
    const float* xptr[NXS];
    int sxoff[NXS];
    #pragma unroll
    for (int ii = 0; ii < NXS; ++ii) {
        const int e = tid + ii * BLOCK;
        if (e < 16 * INPUT_SZ) {
            const int r = e / INPUT_SZ, k = e % INPUT_SZ;
            xptr[ii]  = x + (size_t)(bbase + r) * SEQ * INPUT_SZ + k;  // t = 0
            sxoff[ii] = r * 32 + k;
        } else { xptr[ii] = x; sxoff[ii] = 0; }
    }
    int goff[NEW], haoff[NEW];
    #pragma unroll
    for (int ii = 0; ii < NEW; ++ii) {
        const int e = tid + ii * BLOCK;
        const int rr = (e < 16 * HIDDEN) ? (e / HIDDEN) : 0;
        const int j  = (e < 16 * HIDDEN) ? (e % HIDDEN) : 0;
        goff[ii]  = rr * G4H + j;
        haoff[ii] = rr * 64 + j;
    }
    const char* pf = (const char*)x;
    if (tid < 32)
        pf = (const char*)(x + (size_t)(bbase + (tid & 15)) * SEQ * INPUT_SZ
                             + INPUT_SZ) + (tid >> 4) * 64;   // t = 1
    __syncthreads();   // zero-init + bias visible (needed before bv loads)

    float bv[4];
    #pragma unroll
    for (int q = 0; q < 4; ++q) {
        const int nt = wid + q * NWAVE;
        bv[q] = (nt < NT) ? s_bias[nt * 16 + nloc] : 0.0f;
    }

    // Prologue: stage x for t=0, prefetch t=1
    #pragma unroll
    for (int ii = 0; ii < NXS; ++ii) {
        if (tid + ii * BLOCK < 16 * INPUT_SZ) {
            s_xA[sxoff[ii]] = (_Float16)(*xptr[ii]);
            xptr[ii] += INPUT_SZ;
        }
    }
    if (tid < 32) { __builtin_prefetch(pf, 0, 3); pf += INPUT_SZ * 4; }
    __syncthreads();

    // ================= time loop (2 barriers/step) =================
    #pragma unroll 1
    for (int t = 0; t < SEQ; ++t) {
        // A fragments: lane holds row nloc, K-chunk {half*16..+15}
        const int r = nloc;
        const v16h a_x  = *(const v16h*)&s_xA[r * 32 + half * 16];
        const v16h a_h0 = *(const v16h*)&s_hA[r * 64 + half * 16];
        const v16h a_h1 = *(const v16h*)&s_hA[r * 64 + 32 + half * 16];

        // Tiles wid, wid+4, wid+8: always valid, always n<200 -> straight line
        #pragma unroll
        for (int q = 0; q < 3; ++q) {
            const int nt = wid + q * NWAVE;
            const v8f acc = gate_tile(a_x, a_h0, a_h1,
                                      &s_wihB[(nt * 32 + lane) * 16],
                                      &s_whhB[(nt * 2 * 32 + lane) * 16], bv[q]);
            const int n = nt * 16 + nloc;       // <= 191 < 200 always
            #pragma unroll
            for (int v = 0; v < 8; ++v)
                s_gates[(v + 8 * half) * G4H + n] = acc[v];
        }
        // Tile 12 (n = 192..207): wave 0 only, store-guarded to n<200
        if (wid == 0) {
            const v8f acc = gate_tile(a_x, a_h0, a_h1,
                                      &s_wihB[(12 * 32 + lane) * 16],
                                      &s_whhB[(24 * 32 + lane) * 16], bv[3]);
            const int n = 192 + nloc;
            if (n < G4H) {
                #pragma unroll
                for (int v = 0; v < 8; ++v)
                    s_gates[(v + 8 * half) * G4H + n] = acc[v];
            }
        }
        __syncthreads();   // gates visible; A-frag reads retired (dscnt 0)

        // Cell update (v_tanh_f32) overlapped with staging x(t+1)
        #pragma unroll
        for (int ii = 0; ii < NXS; ++ii) {
            if (t + 1 < SEQ && tid + ii * BLOCK < 16 * INPUT_SZ) {
                s_xA[sxoff[ii]] = (_Float16)(*xptr[ii]);
                xptr[ii] += INPUT_SZ;
            }
        }
        if (t + 2 < SEQ && tid < 32) {
            __builtin_prefetch(pf, 0, 3);
            pf += INPUT_SZ * 4;
        }
        #pragma unroll
        for (int ii = 0; ii < NEW; ++ii) {
            const int e = tid + ii * BLOCK;
            if (e < 16 * HIDDEN) {
                const int g0 = goff[ii];
                const float gi = fast_sigmoid(s_gates[g0]);
                const float gf = fast_sigmoid(s_gates[g0 + HIDDEN]);
                const float gg = fast_tanh   (s_gates[g0 + 2 * HIDDEN]);
                const float go = fast_sigmoid(s_gates[g0 + 3 * HIDDEN]);
                const float c  = gf * s_c[e] + gi * gg;
                s_c[e] = c;
                const float h = go * fast_tanh(c);
                s_h32[e] = h;
                s_hA[haoff[ii]] = (_Float16)h;   // pad cols stay zero
            }
        }
        __syncthreads();   // new h / new x visible for next step
    }

    // ---- final FC: out = h_last @ W_fc^T + b_fc  (16x24, trivial) ----
    #pragma unroll 1
    for (int e = tid; e < 16 * OUT_N; e += BLOCK) {
        const int rr = e / OUT_N, o = e % OUT_N;
        float s = b_fc[o];
        #pragma unroll
        for (int j = 0; j < HIDDEN; ++j)
            s += s_h32[rr * HIDDEN + j] * W_fc[o * HIDDEN + j];
        out[(size_t)(bbase + rr) * OUT_N + o] = s;
    }
}

extern "C" void kernel_launch(void* const* d_in, const int* in_sizes, int n_in,
                              void* d_out, int out_size, void* d_ws, size_t ws_size,
                              hipStream_t stream) {
    const float* x    = (const float*)d_in[0];
    const float* W_ih = (const float*)d_in[1];
    const float* W_hh = (const float*)d_in[2];
    const float* b_ih = (const float*)d_in[3];
    const float* b_hh = (const float*)d_in[4];
    const float* W_fc = (const float*)d_in[5];
    const float* b_fc = (const float*)d_in[6];
    float* out = (float*)d_out;

    dim3 grid(BATCH / 16);   // one block per 16-row batch tile
    dim3 block(BLOCK);       // 4 cooperating wave32s per tile
    hipLaunchKernelGGL(lstm_wmma_kernel, grid, block, 0, stream,
                       x, W_ih, W_hh, b_ih, b_hh, W_fc, b_fc, out);
}
